// Encoder_40149354283321
// MI455X (gfx1250) — compile-verified, matrix operand order checked by reading, not run on hardware
//
#include <hip/hip_runtime.h>
#include <math.h>

// ---------------------------------------------------------------------------
// Problem constants (match reference)
// ---------------------------------------------------------------------------
namespace {
constexpr int BB  = 32;    // batch
constexpr int TT  = 512;   // sequence length
constexpr int CH  = 192;   // channels
constexpr int NH  = 2;     // heads
constexpr int DH  = 96;    // head dim
constexpr int FF  = 768;   // ff channels
constexpr int MEL = 80;
constexpr int DPC = 256;   // duration predictor channels
constexpr int WIN = 4;     // relative attention window
constexpr float EPS  = 1e-4f;
constexpr float SQRT_CH = 13.856406460551018f;   // sqrt(192)
}

// ---------------------------------------------------------------------------
// WMMA types / helpers (CDNA5 gfx1250, wave32)
// ---------------------------------------------------------------------------
typedef __attribute__((ext_vector_type(16))) __bf16 v16bf;
typedef __attribute__((ext_vector_type(8)))  float  v8f;

__device__ __forceinline__ unsigned short f2bf(float f) {
  unsigned int u = __float_as_uint(f);
  u += 0x7FFFu + ((u >> 16) & 1u);          // round-to-nearest-even
  return (unsigned short)(u >> 16);
}
// pack two f32 -> one dword of two bf16 (lo = first element)
__device__ __forceinline__ unsigned int pk2(float lo, float hi) {
  return (unsigned int)f2bf(lo) | ((unsigned int)f2bf(hi) << 16);
}

// Load one 16-element bf16 fragment row (lane m, half) from an LDS row of 32
// bf16 values: k = [half*8, half*8+8) and [16+half*8, ...): two b128 reads.
__device__ __forceinline__ v16bf frag_ld(const unsigned short* rowPtr, int half) {
  const uint4* p = (const uint4*)rowPtr;
  union { uint4 q[2]; v16bf v; } u;
  u.q[0] = p[half];
  u.q[1] = p[2 + half];
  return u.v;
}

#define WMMA_BF16(A, B, C) \
  __builtin_amdgcn_wmma_f32_16x16x32_bf16(false, (A), false, (B), (short)0, (C), false, false)

// ---------------------------------------------------------------------------
// Embedding gather:  y[b,c,t] = emb[x[b,t], c] * sqrt(CH)
// ---------------------------------------------------------------------------
__global__ void __launch_bounds__(256) k_embed(const int* __restrict__ x,
                                               const float* __restrict__ emb,
                                               float* __restrict__ y) {
  int id = blockIdx.x * 256 + threadIdx.x;
  const int n = BB * CH * TT;
  if (id >= n) return;
  int t = id % TT;
  int c = (id / TT) % CH;
  int b = id / (TT * CH);
  int tok = x[b * TT + t];
  y[id] = emb[tok * CH + c] * SQRT_CH;
}

// ---------------------------------------------------------------------------
// conv1d-as-GEMM via WMMA bf16.
//   Y[b,co,t] = bias[co] + sum_{k,ci} W[co,ci,k] * Xin[b,ci,t+k-pad]
// Block tile: 64(co) x 128(t); 8 waves, each wave owns a 32x32 tile (2x2
// WMMAs, fragments reused twice).  CI must be a multiple of 32; CO is guarded
// so any CO works (160 included).
// ---------------------------------------------------------------------------
__global__ void __launch_bounds__(256) k_conv_gemm(
    const float* __restrict__ X, const float* __restrict__ W,
    const float* __restrict__ bias, const float* __restrict__ mask,
    float* __restrict__ Y, int CI, int CO, int KW,
    int relu, int maskX, int maskOut) {
  __shared__ __align__(16) unsigned int Ws32[64 * 16];    // 64co x 32ci bf16
  __shared__ __align__(16) unsigned int Xs32[128 * 16];   // 128t x 32ci bf16
  const unsigned short* Ws = (const unsigned short*)Ws32;
  const unsigned short* Xs = (const unsigned short*)Xs32;
  const int tid  = threadIdx.x;
  const int lane = tid & 31;
  const int wv   = tid >> 5;          // wave 0..7
  const int wc   = wv >> 2;           // 0..1 -> 32-co sub-block
  const int wt   = wv & 3;            // 0..3 -> 32-t  sub-block
  const int m    = lane & 15;
  const int half = lane >> 4;
  const int b   = blockIdx.z;
  const int co0 = blockIdx.y * 64;
  const int t0  = blockIdx.x * 128;
  const int pad = (KW - 1) >> 1;
  const float* __restrict__ Xb   = X + (size_t)b * CI * TT;
  const float* __restrict__ mrow = mask + (size_t)b * TT;

  v8f acc00 = {0.f,0.f,0.f,0.f,0.f,0.f,0.f,0.f};
  v8f acc01 = acc00, acc10 = acc00, acc11 = acc00;

  for (int kk = 0; kk < KW; ++kk) {
    const int sh = kk - pad;
    for (int ci0 = 0; ci0 < CI; ci0 += 32) {
      if (ci0 + 32 < CI)   // pull next weight chunk (global_prefetch_b8)
        __builtin_prefetch(&W[(co0 * CI + ci0 + 32) * KW + kk], 0, 0);
      // --- stage weights: 64co x 32ci as 1024 packed pairs ---
      #pragma unroll
      for (int r = 0; r < 4; ++r) {
        int p  = r * 256 + tid;
        int cp = p & 15, co = p >> 4;
        int cog = co0 + co;
        float w0 = 0.f, w1 = 0.f;
        if (cog < CO) {
          int off = (cog * CI + ci0 + cp * 2) * KW + kk;
          w0 = W[off];
          w1 = W[off + KW];
        }
        Ws32[co * 16 + cp] = pk2(w0, w1);
      }
      // --- stage activations: 128t x 32ci (shifted), 2048 packed pairs ---
      #pragma unroll
      for (int r = 0; r < 8; ++r) {
        int p  = r * 256 + tid;
        int tl = p & 127, cp = p >> 7;   // consecutive lanes -> consecutive t
        int tg = t0 + tl + sh;
        float x0 = 0.f, x1 = 0.f;
        if (tg >= 0 && tg < TT) {
          int off = (ci0 + cp * 2) * TT + tg;
          x0 = Xb[off];
          x1 = Xb[off + TT];
          if (maskX) { float mv = mrow[tg]; x0 *= mv; x1 *= mv; }
        }
        Xs32[tl * 16 + cp] = pk2(x0, x1);
      }
      __syncthreads();
      v16bf a0 = frag_ld(Ws + (wc * 32 + m) * 32, half);
      v16bf a1 = frag_ld(Ws + (wc * 32 + 16 + m) * 32, half);
      v16bf b0 = frag_ld(Xs + (wt * 32 + m) * 32, half);
      v16bf b1 = frag_ld(Xs + (wt * 32 + 16 + m) * 32, half);
      acc00 = WMMA_BF16(a0, b0, acc00);
      acc01 = WMMA_BF16(a0, b1, acc01);
      acc10 = WMMA_BF16(a1, b0, acc10);
      acc11 = WMMA_BF16(a1, b1, acc11);
      __syncthreads();
    }
  }

  // epilogue: bias (+ReLU) (+mask); C layout: lane col = m, rows half*8+r
  float* __restrict__ Yb = Y + (size_t)b * CO * TT;
  #pragma unroll
  for (int j = 0; j < 2; ++j) {
    const int tg = t0 + wt * 32 + j * 16 + m;
    const float mo = maskOut ? mrow[tg] : 1.f;
    #pragma unroll
    for (int i = 0; i < 2; ++i) {
      const v8f acc = (i == 0) ? (j == 0 ? acc00 : acc01)
                               : (j == 0 ? acc10 : acc11);
      #pragma unroll
      for (int r = 0; r < 8; ++r) {
        int co = co0 + wc * 32 + i * 16 + half * 8 + r;
        if (co < CO) {
          float v = acc[r] + bias[co];
          if (relu) v = fmaxf(v, 0.f);
          Yb[co * TT + tg] = v * mo;
        }
      }
    }
  }
}

// ---------------------------------------------------------------------------
// Attention scores: S[bh,t,s] = sum_d (Q[b,h*DH+dd,t]*scale) * K[b,h*DH+dd,s]
// Block tile: 64(t) x 128(s); 8 waves, each a 32x32 tile (2x2 WMMAs).
// ---------------------------------------------------------------------------
__global__ void __launch_bounds__(256) k_attn_scores(
    const float* __restrict__ Q, const float* __restrict__ Km,
    float* __restrict__ S, float scale) {
  __shared__ __align__(16) unsigned int As32[64 * 16];    // [t][d]
  __shared__ __align__(16) unsigned int Bs32[128 * 16];   // [s][d]
  const unsigned short* As = (const unsigned short*)As32;
  const unsigned short* Bs = (const unsigned short*)Bs32;
  const int tid  = threadIdx.x;
  const int lane = tid & 31, wv = tid >> 5;
  const int wm = wv >> 2, wn = wv & 3;
  const int m = lane & 15, half = lane >> 4;
  const int bh = blockIdx.z;
  const int b = bh / NH, h = bh % NH;
  const int t0 = blockIdx.y * 64, s0 = blockIdx.x * 128;
  const float* __restrict__ Qb = Q  + ((size_t)b * CH + h * DH) * TT;
  const float* __restrict__ Kb = Km + ((size_t)b * CH + h * DH) * TT;

  v8f acc00 = {0.f,0.f,0.f,0.f,0.f,0.f,0.f,0.f};
  v8f acc01 = acc00, acc10 = acc00, acc11 = acc00;

  for (int d0 = 0; d0 < DH; d0 += 32) {
    #pragma unroll
    for (int r = 0; r < 4; ++r) {           // 1024 pairs: 64t x 16 d-pairs
      int p  = r * 256 + tid;
      int tl = p & 63, dp = p >> 6;
      int off = (d0 + dp * 2) * TT + t0 + tl;
      As32[tl * 16 + dp] = pk2(Qb[off] * scale, Qb[off + TT] * scale);
    }
    #pragma unroll
    for (int r = 0; r < 8; ++r) {           // 2048 pairs: 128s x 16 d-pairs
      int p  = r * 256 + tid;
      int sl = p & 127, dp = p >> 7;
      int off = (d0 + dp * 2) * TT + s0 + sl;
      Bs32[sl * 16 + dp] = pk2(Kb[off], Kb[off + TT]);
    }
    __syncthreads();
    v16bf a0 = frag_ld(As + (wm * 32 + m) * 32, half);
    v16bf a1 = frag_ld(As + (wm * 32 + 16 + m) * 32, half);
    v16bf b0 = frag_ld(Bs + (wn * 32 + m) * 32, half);
    v16bf b1 = frag_ld(Bs + (wn * 32 + 16 + m) * 32, half);
    acc00 = WMMA_BF16(a0, b0, acc00);
    acc01 = WMMA_BF16(a0, b1, acc01);
    acc10 = WMMA_BF16(a1, b0, acc10);
    acc11 = WMMA_BF16(a1, b1, acc11);
    __syncthreads();
  }

  float* __restrict__ Srow = S + (size_t)bh * TT * TT;
  #pragma unroll
  for (int i = 0; i < 2; ++i) {
    #pragma unroll
    for (int j = 0; j < 2; ++j) {
      const v8f acc = (i == 0) ? (j == 0 ? acc00 : acc01)
                               : (j == 0 ? acc10 : acc11);
      const int sg = s0 + wn * 32 + j * 16 + m;
      #pragma unroll
      for (int r = 0; r < 8; ++r)
        Srow[(t0 + wm * 32 + i * 16 + half * 8 + r) * TT + sg] = acc[r];
    }
  }
}

// ---------------------------------------------------------------------------
// Softmax over s with relative-k window term and mask.
// One block per (b,h,t); 256 threads handle T=512 (2 elems each).
// ---------------------------------------------------------------------------
__global__ void __launch_bounds__(256) k_softmax_rel(
    float* __restrict__ S, const float* __restrict__ Q,
    const float* __restrict__ relk, const float* __restrict__ mask,
    float scale) {
  const int tid = threadIdx.x;
  const int id  = blockIdx.x;
  const int t   = id % TT;
  const int bh  = id / TT;
  const int b = bh / NH, h = bh % NH;
  __shared__ float rel9[16];
  __shared__ float red[256];

  if (tid < 2 * WIN + 1) {                  // rel[t, r] = dot(q[t]*scale, rel_k[r])
    const float* qp = Q + ((size_t)b * CH + h * DH) * TT + t;
    float a = 0.f;
    for (int d = 0; d < DH; ++d) a += qp[d * TT] * relk[tid * DH + d];
    rel9[tid] = a * scale;
  }
  __syncthreads();

  float* __restrict__ row = S + ((size_t)bh * TT + t) * TT;
  const float* __restrict__ mrow = mask + (size_t)b * TT;
  const float mt = mrow[t];
  const int s0 = tid, s1 = tid + 256;
  float v0 = row[s0], v1 = row[s1];
  int r0 = s0 - t; if (r0 >= -WIN && r0 <= WIN) v0 += rel9[r0 + WIN];
  int r1 = s1 - t; if (r1 >= -WIN && r1 <= WIN) v1 += rel9[r1 + WIN];
  if (mt == 0.f || mrow[s0] == 0.f) v0 = -10000.f;
  if (mt == 0.f || mrow[s1] == 0.f) v1 = -10000.f;

  red[tid] = fmaxf(v0, v1);
  __syncthreads();
  for (int off = 128; off; off >>= 1) {
    if (tid < off) red[tid] = fmaxf(red[tid], red[tid + off]);
    __syncthreads();
  }
  const float M = red[0];
  __syncthreads();
  float e0 = __expf(v0 - M), e1 = __expf(v1 - M);
  red[tid] = e0 + e1;
  __syncthreads();
  for (int off = 128; off; off >>= 1) {
    if (tid < off) red[tid] += red[tid + off];
    __syncthreads();
  }
  const float inv = 1.f / red[0];
  row[s0] = e0 * inv;
  row[s1] = e1 * inv;
}

// ---------------------------------------------------------------------------
// out = attn @ V:  O[b, h*DH+dd, t] = sum_s attn[bh,t,s] * V[b,h*DH+dd,s]
// Block tile: 32(d) x 256(t); 8 waves along t, each a 32x32 tile (2x2 WMMAs).
// K-loop over s (512 = 16 chunks of 32); both operands contiguous along s.
// ---------------------------------------------------------------------------
__global__ void __launch_bounds__(256) k_attn_av(
    const float* __restrict__ V, const float* __restrict__ S,
    float* __restrict__ O) {
  __shared__ __align__(16) unsigned int As32[32 * 16];    // [d][s]
  __shared__ __align__(16) unsigned int Bs32[256 * 16];   // [t][s]
  const unsigned short* As = (const unsigned short*)As32;
  const unsigned short* Bs = (const unsigned short*)Bs32;
  const int tid  = threadIdx.x;
  const int lane = tid & 31, wv = tid >> 5;
  const int m = lane & 15, half = lane >> 4;
  const int bh = blockIdx.z;
  const int b = bh / NH, h = bh % NH;
  const int d0 = blockIdx.y * 32;
  const int t0 = blockIdx.x * 256;
  const float* __restrict__ Vb   = V + ((size_t)b * CH + h * DH) * TT;
  const float* __restrict__ Srow = S + (size_t)bh * TT * TT;

  v8f acc00 = {0.f,0.f,0.f,0.f,0.f,0.f,0.f,0.f};
  v8f acc01 = acc00, acc10 = acc00, acc11 = acc00;

  for (int s0 = 0; s0 < TT; s0 += 32) {
    #pragma unroll
    for (int r = 0; r < 2; ++r) {           // 512 pairs: 32d x 16 s-pairs
      int p  = r * 256 + tid;
      int sp = p & 15, dd = p >> 4;
      int off = (d0 + dd) * TT + s0 + sp * 2;
      As32[dd * 16 + sp] = pk2(Vb[off], Vb[off + 1]);
    }
    #pragma unroll
    for (int r = 0; r < 16; ++r) {          // 4096 pairs: 256t x 16 s-pairs
      int p  = r * 256 + tid;
      int sp = p & 15, tl = p >> 4;
      int off = (t0 + tl) * TT + s0 + sp * 2;
      Bs32[tl * 16 + sp] = pk2(Srow[off], Srow[off + 1]);
    }
    __syncthreads();
    v16bf a0 = frag_ld(As + m * 32, half);
    v16bf a1 = frag_ld(As + (16 + m) * 32, half);
    v16bf b0 = frag_ld(Bs + (wv * 32 + m) * 32, half);
    v16bf b1 = frag_ld(Bs + (wv * 32 + 16 + m) * 32, half);
    acc00 = WMMA_BF16(a0, b0, acc00);
    acc01 = WMMA_BF16(a0, b1, acc01);
    acc10 = WMMA_BF16(a1, b0, acc10);
    acc11 = WMMA_BF16(a1, b1, acc11);
    __syncthreads();
  }

  float* __restrict__ Ob = O + ((size_t)b * CH + h * DH) * TT;
  #pragma unroll
  for (int i = 0; i < 2; ++i) {
    #pragma unroll
    for (int j = 0; j < 2; ++j) {
      const v8f acc = (i == 0) ? (j == 0 ? acc00 : acc01)
                               : (j == 0 ? acc10 : acc11);
      const int tg = t0 + wv * 32 + j * 16 + m;
      #pragma unroll
      for (int r = 0; r < 8; ++r) {
        int dd = d0 + i * 16 + half * 8 + r;
        Ob[dd * TT + tg] = acc[r];
      }
    }
  }
}

// ---------------------------------------------------------------------------
// O[b, h*DH+d, t] += sum_{r=-4..4} attn[bh,t,t+r] * rel_v[r+4, d]
// ---------------------------------------------------------------------------
__global__ void __launch_bounds__(256) k_add_relv(
    float* __restrict__ O, const float* __restrict__ S,
    const float* __restrict__ relv) {
  int id = blockIdx.x * 256 + threadIdx.x;
  const int n = BB * NH * DH * TT;
  if (id >= n) return;
  int t = id % TT;
  int d = (id / TT) % DH;
  int h = (id / (TT * DH)) % NH;
  int b = id / (TT * DH * NH);
  const float* arow = S + ((size_t)((b * NH + h) * TT) + t) * TT;
  float a = 0.f;
  #pragma unroll
  for (int r = -WIN; r <= WIN; ++r) {
    int s = t + r;
    if (s >= 0 && s < TT) a += arow[s] * relv[(r + WIN) * DH + d];
  }
  O[((size_t)b * CH + h * DH + d) * TT + t] += a;
}

// ---------------------------------------------------------------------------
// Channel LayerNorm at each (b,t):  Y = LN(X*mk + R) * g + be  (+optional ReLU)
// ---------------------------------------------------------------------------
__global__ void __launch_bounds__(256) k_ln(
    const float* __restrict__ X, const float* __restrict__ R,
    const float* __restrict__ g, const float* __restrict__ be,
    const float* __restrict__ mask, float* __restrict__ Y,
    int C, int relu, int maskX) {
  int id = blockIdx.x * 256 + threadIdx.x;
  if (id >= BB * TT) return;
  int t = id % TT, b = id / TT;
  float mk = maskX ? mask[b * TT + t] : 1.f;
  size_t base = (size_t)b * C * TT + t;
  float mean = 0.f;
  for (int c = 0; c < C; ++c) {
    float v = X[base + c * TT] * mk + (R ? R[base + c * TT] : 0.f);
    mean += v;
  }
  mean /= (float)C;
  float var = 0.f;
  for (int c = 0; c < C; ++c) {
    float v = X[base + c * TT] * mk + (R ? R[base + c * TT] : 0.f) - mean;
    var += v * v;
  }
  var /= (float)C;
  float inv = rsqrtf(var + EPS);
  for (int c = 0; c < C; ++c) {
    float v = X[base + c * TT] * mk + (R ? R[base + c * TT] : 0.f);
    float y = (v - mean) * inv * g[c] + be[c];
    if (relu) y = fmaxf(y, 0.f);
    Y[base + c * TT] = y;
  }
}

// Y = (A + R) * mask   (B,CH,T)
__global__ void __launch_bounds__(256) k_add_mask(
    const float* __restrict__ A, const float* __restrict__ R,
    const float* __restrict__ mask, float* __restrict__ Y) {
  int id = blockIdx.x * 256 + threadIdx.x;
  const int n = BB * CH * TT;
  if (id >= n) return;
  int t = id % TT;
  int b = id / (TT * CH);
  Y[id] = (A[id] + R[id]) * mask[b * TT + t];
}

// log_dur[b,t] = (sum_ci W[ci]*D[b,ci,t]*mk + bias) * mk
__global__ void __launch_bounds__(256) k_dp_final(
    const float* __restrict__ D, const float* __restrict__ Wv,
    const float* __restrict__ bias, const float* __restrict__ mask,
    float* __restrict__ LD) {
  int id = blockIdx.x * 256 + threadIdx.x;
  if (id >= BB * TT) return;
  int t = id % TT, b = id / TT;
  float mk = mask[b * TT + t];
  const float* Db = D + (size_t)b * DPC * TT + t;
  float a = 0.f;
  for (int ci = 0; ci < DPC; ++ci) a += Wv[ci] * Db[ci * TT];
  LD[id] = (a * mk + bias[0]) * mk;
}

// out (B, 161, T): channels 0..159 <- stats, channel 160 <- log_dur
__global__ void __launch_bounds__(256) k_assemble(
    const float* __restrict__ ST, const float* __restrict__ LD,
    float* __restrict__ out) {
  int id = blockIdx.x * 256 + threadIdx.x;
  const int OC = 2 * MEL + 1;
  const int n = BB * OC * TT;
  if (id >= n) return;
  int t = id % TT;
  int c = (id / TT) % OC;
  int b = id / (TT * OC);
  out[id] = (c < 2 * MEL) ? ST[((size_t)b * 2 * MEL + c) * TT + t]
                          : LD[b * TT + t];
}

// ---------------------------------------------------------------------------
// Host-side launcher
// ---------------------------------------------------------------------------
extern "C" void kernel_launch(void* const* d_in, const int* in_sizes, int n_in,
                              void* d_out, int out_size, void* d_ws, size_t ws_size,
                              hipStream_t stream) {
  (void)in_sizes; (void)n_in; (void)out_size; (void)ws_size;
  // Input order: x, mask, embedding, prenet(3 x {w,b,g,be}), pre_w, pre_b,
  // transformer(6 x 18 tensors), proj_w, proj_b, dp(2 x {w,b}), dp_w, dp_b.
  const int*   x     = (const int*)d_in[0];
  const float* mask  = (const float*)d_in[1];
  const float* emb   = (const float*)d_in[2];
  const float* pre_w = (const float*)d_in[15];
  const float* pre_b = (const float*)d_in[16];
  const float* proj_w = (const float*)d_in[125];
  const float* proj_b = (const float*)d_in[126];
  const float* dp0w = (const float*)d_in[127];
  const float* dp0b = (const float*)d_in[128];
  const float* dp1w = (const float*)d_in[129];
  const float* dp1b = (const float*)d_in[130];
  const float* dpw  = (const float*)d_in[131];
  const float* dpb  = (const float*)d_in[132];

  float* ws = (float*)d_ws;
  const size_t SZ = (size_t)BB * CH * TT;
  float* wE  = ws;
  float* wA  = wE + SZ;
  float* wB  = wA + SZ;
  float* wT  = wB + SZ;
  float* wQ  = wT + SZ;
  float* wK  = wQ + SZ;
  float* wV  = wK + SZ;
  float* wO  = wV + SZ;
  float* wH  = wO + SZ;
  float* wFFb = wH + SZ;
  float* wS   = wFFb + (size_t)BB * FF * TT;
  float* wST  = wS + (size_t)BB * NH * TT * TT;
  float* wD0  = wST + (size_t)BB * 2 * MEL * TT;
  float* wD1  = wD0 + (size_t)BB * DPC * TT;
  float* wLD  = wD1 + (size_t)BB * DPC * TT;

  const dim3 blk(256);
  auto ew = [](size_t n) { return dim3((unsigned)((n + 255) / 256)); };
  auto cg = [](int CO) { return dim3(TT / 128, (CO + 63) / 64, BB); };
  const float scale = 0.1020620726159658f;   // 96^-0.5

  // ---- embedding ----
  k_embed<<<ew(SZ), blk, 0, stream>>>(x, emb, wE);

  // ---- prenet: 3 x (conv5 -> LN+ReLU) ----
  const float* cur = wE;
  float* lnout[3] = {wA, wB, wA};
  for (int i = 0; i < 3; ++i) {
    const float* w  = (const float*)d_in[3 + i * 4];
    const float* b  = (const float*)d_in[4 + i * 4];
    const float* g  = (const float*)d_in[5 + i * 4];
    const float* be = (const float*)d_in[6 + i * 4];
    k_conv_gemm<<<cg(CH), blk, 0, stream>>>(
        cur, w, b, mask, wT, CH, CH, 5, /*relu*/0, /*maskX*/1, /*maskOut*/0);
    k_ln<<<ew((size_t)BB * TT), blk, 0, stream>>>(wT, nullptr, g, be, mask,
                                                  lnout[i], CH, /*relu*/1, /*maskX*/0);
    cur = lnout[i];
  }
  // ---- pre 1x1 conv + residual (masked) ----
  k_conv_gemm<<<cg(CH), blk, 0, stream>>>(
      cur, pre_w, pre_b, mask, wT, CH, CH, 1, 0, 0, 0);
  k_add_mask<<<ew(SZ), blk, 0, stream>>>(wT, wE, mask, wH);

  // ---- transformer layers ----
  for (int L = 0; L < 6; ++L) {
    const int base = 17 + L * 18;
    const float* wq   = (const float*)d_in[base + 0];
    const float* bq   = (const float*)d_in[base + 1];
    const float* wk   = (const float*)d_in[base + 2];
    const float* bk   = (const float*)d_in[base + 3];
    const float* wvp  = (const float*)d_in[base + 4];
    const float* bv   = (const float*)d_in[base + 5];
    const float* wo   = (const float*)d_in[base + 6];
    const float* bo   = (const float*)d_in[base + 7];
    const float* relk = (const float*)d_in[base + 8];
    const float* relv = (const float*)d_in[base + 9];
    const float* g0   = (const float*)d_in[base + 10];
    const float* b0   = (const float*)d_in[base + 11];
    const float* w0   = (const float*)d_in[base + 12];
    const float* cb0  = (const float*)d_in[base + 13];
    const float* w1   = (const float*)d_in[base + 14];
    const float* cb1  = (const float*)d_in[base + 15];
    const float* g1   = (const float*)d_in[base + 16];
    const float* b1   = (const float*)d_in[base + 17];

    // QKV (1x1 convs on masked h)
    k_conv_gemm<<<cg(CH), blk, 0, stream>>>(wH, wq, bq, mask, wQ, CH, CH, 1, 0, 1, 0);
    k_conv_gemm<<<cg(CH), blk, 0, stream>>>(wH, wk, bk, mask, wK, CH, CH, 1, 0, 1, 0);
    k_conv_gemm<<<cg(CH), blk, 0, stream>>>(wH, wvp, bv, mask, wV, CH, CH, 1, 0, 1, 0);

    // scores = (q*scale) @ k^T, + rel-k window, masked softmax
    k_attn_scores<<<dim3(TT / 128, TT / 64, BB * NH), blk, 0, stream>>>(
        wQ, wK, wS, scale);
    k_softmax_rel<<<dim3(BB * NH * TT), blk, 0, stream>>>(wS, wQ, relk, mask, scale);

    // out = attn @ v  (+ rel-v window term)
    k_attn_av<<<dim3(TT / 256, DH / 32, BB * NH), blk, 0, stream>>>(wV, wS, wO);
    k_add_relv<<<ew((size_t)BB * NH * DH * TT), blk, 0, stream>>>(wO, wS, relv);

    // output projection, LN(x + res)
    k_conv_gemm<<<cg(CH), blk, 0, stream>>>(wO, wo, bo, mask, wT, CH, CH, 1, 0, 0, 0);
    k_ln<<<ew((size_t)BB * TT), blk, 0, stream>>>(wT, wH, g0, b0, mask, wA, CH, 0, 0);
    // FF: conv3 192->768 (ReLU), conv3 768->192, LN(x*mask + res)
    k_conv_gemm<<<cg(FF), blk, 0, stream>>>(wA, w0, cb0, mask, wFFb, CH, FF, 3, 1, 1, 0);
    k_conv_gemm<<<cg(CH), blk, 0, stream>>>(wFFb, w1, cb1, mask, wT, FF, CH, 3, 0, 1, 0);
    k_ln<<<ew((size_t)BB * TT), blk, 0, stream>>>(wT, wA, g1, b1, mask, wH, CH, 0, 1);
  }

  // ---- projection to mean/log_std (masked in and out) ----
  k_conv_gemm<<<cg(2 * MEL), blk, 0, stream>>>(
      wH, proj_w, proj_b, mask, wST, CH, 2 * MEL, 1, 0, 1, 1);

  // ---- duration predictor ----
  k_conv_gemm<<<cg(DPC), blk, 0, stream>>>(wH, dp0w, dp0b, mask, wD0, CH, DPC, 3, 1, 1, 0);
  k_conv_gemm<<<cg(DPC), blk, 0, stream>>>(wD0, dp1w, dp1b, mask, wD1, DPC, DPC, 3, 1, 1, 0);
  k_dp_final<<<ew((size_t)BB * TT), blk, 0, stream>>>(wD1, dpw, dpb, mask, wLD);

  // ---- assemble output (B, 161, T) ----
  k_assemble<<<ew((size_t)BB * (2 * MEL + 1) * TT), blk, 0, stream>>>(
      wST, wLD, (float*)d_out);
}
